// SLOT_SEG_25280177504748
// MI455X (gfx1250) — matrix operand order repaired and checked by reading, not compiled, not run on hardware
//
#include <hip/hip_runtime.h>
#include <hip/hip_bf16.h>

// ---------------- CDNA5 WMMA types/helpers ----------------
typedef __attribute__((ext_vector_type(16))) __bf16 v16bf;
typedef __attribute__((ext_vector_type(8)))  float  v8f;
typedef __attribute__((address_space(1))) float  gfloat;   // explicit global AS
typedef __attribute__((address_space(1))) __bf16 gbf;

#define WMMA_BF16(a, b, c) \
  __builtin_amdgcn_wmma_f32_16x16x32_bf16(false, (a), false, (b), (short)0, (c), false, false)

static __device__ __forceinline__ v8f zero8() {
  v8f c;
#pragma unroll
  for (int i = 0; i < 8; ++i) c[i] = 0.0f;
  return c;
}

// A fragment: 16x32 bf16, row m = lane&15; halves 0..7 hold K=kb..kb+7,
// halves 8..15 hold K=kb+16..kb+23, kb = (lane>=16)?8:0  (ISA 7.12.2)
template <typename PT>
static __device__ __forceinline__ v16bf load_a_frag_t(PT src, int ld, int m0, int k0,
                                                      int lane) {
  int m = lane & 15;
  int kb = (lane >> 4) << 3;
  PT p = src + (m0 + m) * ld + k0 + kb;
  v16bf a;
#pragma unroll
  for (int h = 0; h < 8; ++h) { a[h] = p[h]; a[8 + h] = p[16 + h]; }
  return a;
}

// B fragment: 32x16 bf16, col n = lane&15; halves 0..7 hold K=kb..kb+7,
// halves 8..15 hold K=kb+16..kb+23, kb = (lane>=16)?8:0. Element (k,n) at src[(n0+n)*sN + (k0+k)*sK].
template <typename PT>
static __device__ __forceinline__ v16bf load_b_frag_t(PT src, int sN, int sK, int n0,
                                                      int k0, int lane) {
  int n = lane & 15;
  int kb = (lane >> 4) << 3;
  PT p = src + (n0 + n) * sN + (k0 + kb) * sK;
  v16bf b;
#pragma unroll
  for (int h = 0; h < 8; ++h) { b[h] = p[h * sK]; b[8 + h] = p[(16 + h) * sK]; }
  return b;
}

// LayerNorm of one 256-wide row by one wave32; write bf16
static __device__ __forceinline__ void ln_row_to_bf(const float* src, __bf16* dst,
                                                    const float* g, const float* bvec,
                                                    int lane) {
  float s = 0.f, s2 = 0.f;
#pragma unroll
  for (int k = lane; k < 256; k += 32) { float v = src[k]; s += v; s2 += v * v; }
#pragma unroll
  for (int m = 16; m >= 1; m >>= 1) { s += __shfl_xor(s, m, 32); s2 += __shfl_xor(s2, m, 32); }
  float mean = s * (1.0f / 256.0f);
  float var  = s2 * (1.0f / 256.0f) - mean * mean;
  float rstd = rsqrtf(var + 1e-5f);
#pragma unroll
  for (int k = lane; k < 256; k += 32)
    dst[k] = (__bf16)((src[k] - mean) * rstd * g[k] + bvec[k]);
}

static __device__ __forceinline__ void ln_row_to_f32(const float* src, float* dst,
                                                     const float* g, const float* bvec,
                                                     int lane) {
  float s = 0.f, s2 = 0.f;
#pragma unroll
  for (int k = lane; k < 256; k += 32) { float v = src[k]; s += v; s2 += v * v; }
#pragma unroll
  for (int m = 16; m >= 1; m >>= 1) { s += __shfl_xor(s, m, 32); s2 += __shfl_xor(s2, m, 32); }
  float mean = s * (1.0f / 256.0f);
  float var  = s2 * (1.0f / 256.0f) - mean * mean;
  float rstd = rsqrtf(var + 1e-5f);
#pragma unroll
  for (int k = lane; k < 256; k += 32)
    dst[k] = (src[k] - mean) * rstd * g[k] + bvec[k];
}

// Paired-N tile GEMM: A (LDS bf16, ld=256) @ Wglobal^T (+bias).
// Each wave owns PAIRS (mt, nt0/nt0+1) tile pairs; one A fragment feeds 2 WMMAs.
// MODE 0: f32 store; MODE 1: relu->bf16; MODE 2: bf16; MODE 3: bf16 transposed (dim-major, ld 64)
template <int MODE, int PAIRS>
static __device__ void gemm_tiles2(const __bf16* A, const gbf* W, const float* bias,
                                   float* dstF, __bf16* dstB, int wave, int lane) {
  int n = lane & 15, mo = (lane >> 4) << 3;
#pragma unroll 1
  for (int p = 0; p < PAIRS; ++p) {
    int pi = wave + 8 * p;
    int mt = pi >> 3, ntp = pi & 7;
    int nt0 = ntp * 2, nt1 = nt0 + 1;
    v8f c0 = zero8(), c1 = zero8();
#pragma unroll
    for (int k0 = 0; k0 < 256; k0 += 32) {
      v16bf a = load_a_frag_t<const __bf16*>(A, 256, mt * 16, k0, lane);
      v16bf b0 = load_b_frag_t<const gbf*>(W, 256, 1, nt0 * 16, k0, lane);
      v16bf b1 = load_b_frag_t<const gbf*>(W, 256, 1, nt1 * 16, k0, lane);
      c0 = WMMA_BF16(a, b0, c0);
      c1 = WMMA_BF16(a, b1, c1);
    }
#pragma unroll
    for (int half = 0; half < 2; ++half) {
      v8f& c = half ? c1 : c0;
      int col = (half ? nt1 : nt0) * 16 + n;
      float bv = bias[col];
#pragma unroll
      for (int rr = 0; rr < 8; ++rr) {
        int row = mt * 16 + mo + rr;
        float v = c[rr] + bv;
        if (MODE == 0)      dstF[row * 256 + col] = v;
        else if (MODE == 1) dstB[row * 256 + col] = (__bf16)fmaxf(v, 0.0f);
        else if (MODE == 2) dstB[row * 256 + col] = (__bf16)v;
        else                dstB[col * 64 + row] = (__bf16)v;   // transposed
      }
    }
  }
}

// ---------------- prep kernels ----------------
__global__ void cvt_bf16_kernel(const float* __restrict__ src, __bf16* __restrict__ dst, int n) {
  int i = blockIdx.x * 256 + threadIdx.x;
  if (i < n) dst[i] = (__bf16)src[i];
}

__global__ void pe_kernel(const float* __restrict__ pe_w, const float* __restrict__ pe_b,
                          float* __restrict__ ws_pe) {
  int t = blockIdx.x;    // 0..767 -> (fi, y, x)
  int d = threadIdx.x;   // 0..255
  int fi = t / 192, rem = t % 192, y = rem / 24, xw = rem % 24;
  float c0 = fi * (1.0f / 3.0f), c1 = y * (1.0f / 7.0f), c2 = xw * (1.0f / 23.0f);
  const float* w = pe_w + d * 6;
  ws_pe[t * 256 + d] = pe_b[d] + c0 * w[0] + c1 * w[1] + c2 * w[2] +
                       (1.f - c0) * w[3] + (1.f - c1) * w[4] + (1.f - c2) * w[5];
}

__global__ void slotsn_kernel(const float* __restrict__ slots, const float* ns_g,
                              const float* ns_b, float* __restrict__ slotsn,
                              __bf16* __restrict__ qbf) {
  int tid = threadIdx.x, wave = tid >> 5, lane = tid & 31;
#pragma unroll 1
  for (int rt = 0; rt < 3; ++rt) {
    int row = wave + 8 * rt;
    if (row < 22) ln_row_to_f32(slots + row * 256, slotsn + row * 256, ns_g, ns_b, lane);
  }
  for (int idx = tid; idx < 10 * 256; idx += 256) qbf[22 * 256 + idx] = (__bf16)0.0f;
}

__global__ void qhg_kernel(const float* __restrict__ slotsn, const float* __restrict__ q_w,
                           const float* q_b, const float* __restrict__ slots,
                           const float* __restrict__ gru_wh, const float* gru_bh,
                           __bf16* __restrict__ qbf, float* __restrict__ hg) {
  int idx = blockIdx.x * 256 + threadIdx.x;
  if (idx < 22 * 256) {
    int i = idx >> 8, n = idx & 255;
    float acc = q_b[n];
    const float* s = slotsn + i * 256;
    const float* w = q_w + n * 256;
    for (int k = 0; k < 256; ++k) acc += s[k] * w[k];
    qbf[i * 256 + n] = (__bf16)acc;
  } else if (idx < 22 * 256 + 22 * 768) {
    int id2 = idx - 22 * 256;
    int i = id2 / 768, n = id2 % 768;
    float acc = gru_bh[n];
    const float* s = slots + i * 256;
    const float* w = gru_wh + n * 256;
    for (int k = 0; k < 256; ++k) acc += s[k] * w[k];
    hg[i * 768 + n] = acc;
  }
}

// ---------------- main fused slot-attention kernel ----------------
// one block per (window,batch); 256 threads = 8 wave32s; 12 chunks of 64 tokens
// LDS ~220KB -> 1 workgroup/WGP (2 waves/SIMD): launch_bounds(256,2) gives the
// allocator half the register file per wave.
__global__ __launch_bounds__(256, 2) void slot_main(
    const float* __restrict__ in_, const float* __restrict__ slots,
    const float* LN_g, const float* LN_b, const float* ni_g, const float* ni_b,
    const float* npf_g, const float* npf_b,
    const float* FC1_b, const float* FC2_b, const float* k_b, const float* v_b,
    const float* ff1_b, const float* ff2_b, const float* gru_bi,
    const float* __restrict__ ws_pe_,
    const __bf16* __restrict__ wFC1_, const __bf16* __restrict__ wFC2_,
    const __bf16* __restrict__ wK_, const __bf16* __restrict__ wV_,
    const __bf16* __restrict__ wFF1_, const __bf16* __restrict__ wFF2_,
    const __bf16* __restrict__ wGRU_, const __bf16* __restrict__ qbf_,
    const float* __restrict__ hg_, float* __restrict__ out_) {
  // explicit global address-space views -> global_load/store (not FLAT)
  const gfloat* in    = (const gfloat*)in_;
  const gfloat* ws_pe = (const gfloat*)ws_pe_;
  const gfloat* hg    = (const gfloat*)hg_;
  const gbf* wFC1 = (const gbf*)wFC1_;
  const gbf* wFC2 = (const gbf*)wFC2_;
  const gbf* wK   = (const gbf*)wK_;
  const gbf* wV   = (const gbf*)wV_;
  const gbf* wFF1 = (const gbf*)wFF1_;
  const gbf* wFF2 = (const gbf*)wFF2_;
  const gbf* wGRU = (const gbf*)wGRU_;
  const gbf* qbf  = (const gbf*)qbf_;
  gfloat* out = (gfloat*)out_;

  extern __shared__ char smem[];
  float*  sX    = (float*)(smem);                 // 64*256 f32  (65536 B)
  __bf16* sA    = (__bf16*)(smem + 65536);        // 64*256 bf16 (32768 B)
  __bf16* sKc   = (__bf16*)(smem + 98304);        // 64*256 bf16 (token-major)
  __bf16* sVc   = (__bf16*)(smem + 131072);       // 64*256 bf16 (FC1 scratch row-major; v dim-major)
  float*  sDots = (float*)(smem + 163840);        // 32*64 f32   (8192 B)
  __bf16* sAt   = (__bf16*)(smem + 172032);       // 32*64 bf16  (4096 B)
  float*  sNum  = (float*)(smem + 176128);        // 32*256 f32  (32768 B) num accumulator
  float*  sDen  = (float*)(smem + 208896);        // 32 f32      (128 B)
  __bf16* sQ    = (__bf16*)(smem + 209024);       // 32*256 bf16 (16384 B) -> 225408 total

  int tid = threadIdx.x, wave = tid >> 5, lane = tid & 31;
  int bb = blockIdx.x;
  int wwin = bb >> 5, batch = bb & 31;
  const long in_base = (long)batch * (16L * 8 * 24 * 256) + (long)(2 * wwin) * (8L * 24 * 256);
  gfloat* attn_out = out + 1146880L + (long)(batch * 7 + wwin) * 22 * 768;
  gfloat* slot_out = out + (long)(batch * 7 + wwin) * 20 * 256;
  int n15 = lane & 15, mo = (lane >> 4) << 3;

  // init: zero attn pad rows (22..31), zero den + num, stage q into LDS
  for (int idx = tid; idx < 10 * 64; idx += 256) sAt[22 * 64 + idx] = (__bf16)0.0f;
  if (tid < 32) sDen[tid] = 0.0f;
  for (int idx = tid; idx < 32 * 256; idx += 256) {
    sNum[idx] = 0.0f;
    sQ[idx] = qbf[idx];
  }
  __syncthreads();

#pragma unroll 1
  for (int ch = 0; ch < 12; ++ch) {
    int t0 = ch * 64;
    // stage 0: x = input + positional embedding
    for (int idx = tid; idx < 64 * 256; idx += 256) {
      int tl = idx >> 8, d = idx & 255;
      long tg = t0 + tl;
      sX[idx] = in[in_base + tg * 256 + d] + ws_pe[tg * 256 + d];
    }
    if (ch + 1 < 12)  // global_prefetch_b8 of next chunk
      __builtin_prefetch(in_ + in_base + (long)(t0 + 64) * 256 + tid * 64, 0, 3);
    __syncthreads();
    // stage 1: LN -> bf16
#pragma unroll 1
    for (int rt = 0; rt < 8; ++rt)
      ln_row_to_bf(sX + (wave * 8 + rt) * 256, sA + (wave * 8 + rt) * 256, LN_g, LN_b, lane);
    __syncthreads();
    // FC1 + relu (bf16 scratch in sVc, row-major)
    gemm_tiles2<1, 4>(sA, wFC1, FC1_b, nullptr, sVc, wave, lane);
    __syncthreads();
    // FC2 -> f32 sX
    gemm_tiles2<0, 4>(sVc, wFC2, FC2_b, sX, nullptr, wave, lane);
    __syncthreads();
    // norm_input -> bf16 sA
#pragma unroll 1
    for (int rt = 0; rt < 8; ++rt)
      ln_row_to_bf(sX + (wave * 8 + rt) * 256, sA + (wave * 8 + rt) * 256, ni_g, ni_b, lane);
    __syncthreads();
    // k (token-major) and v (dim-major / transposed) projections
    gemm_tiles2<2, 4>(sA, wK, k_b, nullptr, sKc, wave, lane);
    gemm_tiles2<3, 4>(sA, wV, v_b, nullptr, sVc, wave, lane);
    __syncthreads();
    // dots = q @ k^T * 1/16  (M=32 pad, N=64 tokens, K=256); 1 tile per wave
    {
      int mt = wave >> 2, nt = wave & 3;
      v8f c = zero8();
#pragma unroll
      for (int k0 = 0; k0 < 256; k0 += 32) {
        v16bf a = load_a_frag_t<const __bf16*>(sQ, 256, mt * 16, k0, lane);
        v16bf b = load_b_frag_t<const __bf16*>(sKc, 256, 1, nt * 16, k0, lane);
        c = WMMA_BF16(a, b, c);
      }
      float* dbase = sDots + (mt * 16 + mo) * 64 + nt * 16 + n15;
#pragma unroll
      for (int rr = 0; rr < 8; ++rr) dbase[rr * 64] = c[rr] * 0.0625f;
    }
    __syncthreads();
    // softmax over slot axis per token; write attn_ori (+EPS) to output + LDS
    if (tid < 64) {
      int j = tid;
      float mx = -1e30f;
      for (int i = 0; i < 22; ++i) mx = fmaxf(mx, sDots[i * 64 + j]);
      float sum = 0.f;
      for (int i = 0; i < 22; ++i) sum += __expf(sDots[i * 64 + j] - mx);
      float rs = 1.0f / sum;
      for (int i = 0; i < 22; ++i) {
        float ao = __expf(sDots[i * 64 + j] - mx) * rs + 1e-8f;
        sDots[i * 64 + j] = ao;
        sAt[i * 64 + j] = (__bf16)ao;
        attn_out[(long)i * 768 + t0 + j] = ao;
      }
    }
    __syncthreads();
    // den accumulation (row sums over tokens)
    if (tid < 22) {
      float s = 0.f;
      for (int j = 0; j < 64; ++j) s += sDots[tid * 64 + j];
      sDen[tid] += s;
    }
    // num += attn_ori @ vv  (M=32 pad, N=256, K=64); accumulate into LDS via
    // ds_add_f32. ONE base address + compile-time immediate offsets
    // (rr*1024 + q*256 bytes) so no per-element address registers survive
    // across the chunk loop.
    {
      int mt = wave & 1;
      v8f cn4[4];
#pragma unroll
      for (int q = 0; q < 4; ++q) cn4[q] = zero8();
#pragma unroll
      for (int k0 = 0; k0 < 64; k0 += 32) {
        v16bf a = load_a_frag_t<const __bf16*>(sAt, 64, mt * 16, k0, lane);
#pragma unroll
        for (int q = 0; q < 4; ++q) {
          int nt = (wave >> 1) + 4 * q;
          // vv stored dim-major: element (k,n) = sVc[n*64 + k] -> contiguous K
          v16bf b = load_b_frag_t<const __bf16*>(sVc, 64, 1, nt * 16, k0, lane);
          cn4[q] = WMMA_BF16(a, b, cn4[q]);
        }
      }
      float* nbase = sNum + (mt * 16 + mo) * 256 + (wave >> 1) * 16 + n15;
#pragma unroll
      for (int q = 0; q < 4; ++q)
#pragma unroll
        for (int rr = 0; rr < 8; ++rr)
          atomicAdd(nbase + rr * 256 + q * 64, cn4[q][rr]);
    }
    __syncthreads();
  }

  // upd = num / den -> bf16 (pad rows zero), reuse sA
  for (int idx = tid; idx < 32 * 256; idx += 256) {
    int i = idx >> 8;
    sA[idx] = (__bf16)((i < 22) ? sNum[idx] / sDen[i] : 0.0f);
  }
  __syncthreads();
  // GRU: xg = upd @ gru_wi^T (gates r,z,n fused per (mt,dt) pair; one A -> 3 WMMAs)
  {
    int mt = wave & 1;
#pragma unroll 1
    for (int q = 0; q < 4; ++q) {
      int dt = (wave >> 1) + 4 * q;
      v8f cr = zero8(), cz = zero8(), cn = zero8();
#pragma unroll
      for (int k0 = 0; k0 < 256; k0 += 32) {
        v16bf a = load_a_frag_t<const __bf16*>(sA, 256, mt * 16, k0, lane);
        cr = WMMA_BF16(a, load_b_frag_t<const gbf*>(wGRU, 256, 1, 0 * 256 + dt * 16, k0, lane), cr);
        cz = WMMA_BF16(a, load_b_frag_t<const gbf*>(wGRU, 256, 1, 1 * 256 + dt * 16, k0, lane), cz);
        cn = WMMA_BF16(a, load_b_frag_t<const gbf*>(wGRU, 256, 1, 2 * 256 + dt * 16, k0, lane), cn);
      }
      int d = dt * 16 + n15;
#pragma unroll
      for (int rr = 0; rr < 8; ++rr) {
        int i = mt * 16 + mo + rr;
        if (i < 20) {
          float ir = cr[rr] + gru_bi[d];
          float iz = cz[rr] + gru_bi[256 + d];
          float in_g = cn[rr] + gru_bi[512 + d];
          float hr = hg[i * 768 + d], hz = hg[i * 768 + 256 + d], hn = hg[i * 768 + 512 + d];
          float rg = 1.0f / (1.0f + __expf(-(ir + hr)));
          float zg = 1.0f / (1.0f + __expf(-(iz + hz)));
          float ng = tanhf(in_g + rg * hn);
          sNum[i * 256 + d] = (1.0f - zg) * ng + zg * slots[i * 256 + d];
        }
      }
    }
  }
  __syncthreads();
  // residual MLP: LN -> ff1(relu) -> ff2 + residual
#pragma unroll 1
  for (int rt = 0; rt < 3; ++rt) {
    int row = wave + 8 * rt;
    if (row < 20) ln_row_to_bf(sNum + row * 256, sA + row * 256, npf_g, npf_b, lane);
  }
  for (int idx = tid; idx < 12 * 256; idx += 256) sA[20 * 256 + idx] = (__bf16)0.0f;
  __syncthreads();
  gemm_tiles2<1, 2>(sA, wFF1, ff1_b, nullptr, sVc, wave, lane);
  __syncthreads();
  // ff2 + residual -> final slots_out (paired-N tiles)
  {
#pragma unroll 1
    for (int p = 0; p < 2; ++p) {
      int pi = wave + 8 * p;
      int mt = pi >> 3, ntp = pi & 7;
      int nt0 = ntp * 2, nt1 = nt0 + 1;
      v8f c0 = zero8(), c1 = zero8();
#pragma unroll
      for (int k0 = 0; k0 < 256; k0 += 32) {
        v16bf a = load_a_frag_t<const __bf16*>(sVc, 256, mt * 16, k0, lane);
        v16bf b0 = load_b_frag_t<const gbf*>(wFF2, 256, 1, nt0 * 16, k0, lane);
        v16bf b1 = load_b_frag_t<const gbf*>(wFF2, 256, 1, nt1 * 16, k0, lane);
        c0 = WMMA_BF16(a, b0, c0);
        c1 = WMMA_BF16(a, b1, c1);
      }
#pragma unroll
      for (int half = 0; half < 2; ++half) {
        v8f& c = half ? c1 : c0;
        int col = (half ? nt1 : nt0) * 16 + n15;
        float bv = ff2_b[col];
#pragma unroll
        for (int rr = 0; rr < 8; ++rr) {
          int row = mt * 16 + mo + rr;
          if (row < 20)
            slot_out[row * 256 + col] = c[rr] + bv + sNum[row * 256 + col];
        }
      }
    }
  }
}

// ---------------- host launcher ----------------
extern "C" void kernel_launch(void* const* d_in, const int* in_sizes, int n_in,
                              void* d_out, int out_size, void* d_ws, size_t ws_size,
                              hipStream_t stream) {
  const float* in     = (const float*)d_in[0];
  const float* slots  = (const float*)d_in[1];
  const float* pe_w   = (const float*)d_in[2];
  const float* pe_b   = (const float*)d_in[3];
  const float* LN_g   = (const float*)d_in[4];
  const float* LN_b   = (const float*)d_in[5];
  const float* ni_g   = (const float*)d_in[6];
  const float* ni_b   = (const float*)d_in[7];
  const float* ns_g   = (const float*)d_in[8];
  const float* ns_b   = (const float*)d_in[9];
  const float* npf_g  = (const float*)d_in[10];
  const float* npf_b  = (const float*)d_in[11];
  const float* FC1_w  = (const float*)d_in[12];
  const float* FC1_b  = (const float*)d_in[13];
  const float* FC2_w  = (const float*)d_in[14];
  const float* FC2_b  = (const float*)d_in[15];
  const float* q_w    = (const float*)d_in[16];
  const float* q_b    = (const float*)d_in[17];
  const float* k_w    = (const float*)d_in[18];
  const float* k_b    = (const float*)d_in[19];
  const float* v_w    = (const float*)d_in[20];
  const float* v_b    = (const float*)d_in[21];
  const float* ff1_w  = (const float*)d_in[22];
  const float* ff1_b  = (const float*)d_in[23];
  const float* ff2_w  = (const float*)d_in[24];
  const float* ff2_b  = (const float*)d_in[25];
  const float* gru_wi = (const float*)d_in[26];
  const float* gru_wh = (const float*)d_in[27];
  const float* gru_bi = (const float*)d_in[28];
  const float* gru_bh = (const float*)d_in[29];

  char* ws = (char*)d_ws;
  float*  ws_pe     = (float*)ws;   ws += 768 * 256 * 4;     // 786432 B
  float*  ws_slotsn = (float*)ws;   ws += 22 * 256 * 4;      // 22528 B
  __bf16* ws_qbf    = (__bf16*)ws;  ws += 32 * 256 * 2;      // 16384 B
  float*  ws_hg     = (float*)ws;   ws += 22 * 768 * 4;      // 67584 B
  __bf16* wFC1      = (__bf16*)ws;  ws += 256 * 256 * 2;
  __bf16* wFC2      = (__bf16*)ws;  ws += 256 * 256 * 2;
  __bf16* wK        = (__bf16*)ws;  ws += 256 * 256 * 2;
  __bf16* wV        = (__bf16*)ws;  ws += 256 * 256 * 2;
  __bf16* wFF1      = (__bf16*)ws;  ws += 256 * 256 * 2;
  __bf16* wFF2      = (__bf16*)ws;  ws += 256 * 256 * 2;
  __bf16* wGRU      = (__bf16*)ws;  ws += 768 * 256 * 2;     // total ~2.1 MB

  cvt_bf16_kernel<<<256, 256, 0, stream>>>(FC1_w, wFC1, 65536);
  cvt_bf16_kernel<<<256, 256, 0, stream>>>(FC2_w, wFC2, 65536);
  cvt_bf16_kernel<<<256, 256, 0, stream>>>(k_w, wK, 65536);
  cvt_bf16_kernel<<<256, 256, 0, stream>>>(v_w, wV, 65536);
  cvt_bf16_kernel<<<256, 256, 0, stream>>>(ff1_w, wFF1, 65536);
  cvt_bf16_kernel<<<256, 256, 0, stream>>>(ff2_w, wFF2, 65536);
  cvt_bf16_kernel<<<768, 256, 0, stream>>>(gru_wi, wGRU, 196608);
  pe_kernel<<<768, 256, 0, stream>>>(pe_w, pe_b, ws_pe);
  slotsn_kernel<<<1, 256, 0, stream>>>(slots, ns_g, ns_b, ws_slotsn, ws_qbf);
  qhg_kernel<<<88, 256, 0, stream>>>(ws_slotsn, q_w, q_b, slots, gru_wh, gru_bh,
                                     ws_qbf, ws_hg);

  size_t shmem = 225408;  // within CDNA5's 320 KB per-workgroup LDS
  slot_main<<<224, 256, shmem, stream>>>(
      in, slots, LN_g, LN_b, ni_g, ni_b, npf_g, npf_b,
      FC1_b, FC2_b, k_b, v_b, ff1_b, ff2_b, gru_bi,
      ws_pe, wFC1, wFC2, wK, wV, wFF1, wFF2, wGRU, ws_qbf, ws_hg,
      (float*)d_out);
  (void)in_sizes; (void)n_in; (void)out_size; (void)ws_size;
}
